// NanoGPT_18889266168447
// MI455X (gfx1250) — matrix operand order repaired and checked by reading, not compiled, not run on hardware
//
// NanoGPT forward for MI455X (gfx1250, wave32, WMMA + async global->LDS).
//
// Roofline: ~2.6 TFLOP of GEMM vs ~3.8 GB of HBM traffic (dominated by the
// 1.65 GB logits tensor) => compute-bound; all matmuls routed through
// v_wmma_f32_16x16x32_bf16 (bf16 in / f32 accum keeps fp32 exponent range for
// the 0.02-scale weights). Weights converted f32->bf16 *transposed* (N-major)
// once per layer so B-side WMMA fragments load contiguously in K from LDS.
// GEMM tiles are staged with GLOBAL_LOAD_ASYNC_TO_LDS_B128 (ASYNCcnt) into a
// double-buffered LDS so next-tile fills overlap the WMMA work.

#include <hip/hip_runtime.h>
#include <hip/hip_bf16.h>

#define DEVFN __device__ __forceinline__

typedef __attribute__((ext_vector_type(16))) __bf16 v16bf;
typedef __attribute__((ext_vector_type(8)))  float  v8f;
typedef __attribute__((ext_vector_type(4)))  int    ivec4;
typedef int avec4 __attribute__((vector_size(16)));   // builtin's 16B unit

union FragBF {
  struct { ivec4 lo; ivec4 hi; } i;
  v16bf v;
};

static constexpr int LNUM = 12, CDIM = 768, NH = 12, HSZ = 64, FDIM = 3072;
static constexpr int VOCAB = 50257, BB = 8, TT = 1024, BT = BB * TT;

DEVFN unsigned short f32_to_bf16(float f) {
  unsigned u = __float_as_uint(f);
  unsigned r = u + 0x7fffu + ((u >> 16) & 1u);   // round-to-nearest-even
  return (unsigned short)(r >> 16);
}

DEVFN v8f wmma_bf16(v16bf a, v16bf b, v8f c) {
  return __builtin_amdgcn_wmma_f32_16x16x32_bf16(false, a, false, b,
                                                 (short)0, c, false, false);
}

// ---- async global->LDS (CDNA5); fall back to register staging if absent ----
#if __has_builtin(__builtin_amdgcn_global_load_async_to_lds_b128) && \
    __has_builtin(__builtin_amdgcn_s_wait_asynccnt)
#define ASYNC_LDS 1
#else
#define ASYNC_LDS 0
#endif

DEVFN void cp_async16(const unsigned short* g, unsigned short* l) {
#if ASYNC_LDS
  __builtin_amdgcn_global_load_async_to_lds_b128(
      (__attribute__((address_space(1))) avec4*)g,
      (__attribute__((address_space(3))) avec4*)l, 0, 0);
#else
  *(ivec4*)l = *(const ivec4*)g;
#endif
}

DEVFN void async_wait_all() {
#if ASYNC_LDS
  __builtin_amdgcn_s_wait_asynccnt(0);
#endif
}

// ---------------------------------------------------------------- embedding
__global__ void embed_kernel(const int* __restrict__ inp,
                             const float* __restrict__ tok,
                             const float* __restrict__ pos,
                             float* __restrict__ x) {
  int row = blockIdx.x;                // 0..BT-1
  int t   = row % TT;
  int tkn = inp[row];
  const float* tr = tok + (size_t)tkn * CDIM;
  const float* pr = pos + (size_t)t * CDIM;
  float* xr = x + (size_t)row * CDIM;
  for (int c = threadIdx.x; c < CDIM; c += 256) xr[c] = tr[c] + pr[c];
}

// ---------------------------------------------------------------- layernorm
__global__ void ln_kernel(const float* __restrict__ x,
                          const float* __restrict__ g,
                          const float* __restrict__ b,
                          unsigned short* __restrict__ out) {
  __shared__ float smem[16];
  int row = blockIdx.x;
  const float* xr = x + (size_t)row * CDIM;
  float s = 0.f, s2 = 0.f;
  for (int c = threadIdx.x; c < CDIM; c += 256) {
    float v = xr[c]; s += v; s2 += v * v;
  }
  for (int m = 16; m >= 1; m >>= 1) {
    s  += __shfl_xor(s,  m, 32);
    s2 += __shfl_xor(s2, m, 32);
  }
  int w = threadIdx.x >> 5;
  if ((threadIdx.x & 31) == 0) { smem[w] = s; smem[8 + w] = s2; }
  __syncthreads();
  float ts = 0.f, ts2 = 0.f;
  for (int i = 0; i < 8; i++) { ts += smem[i]; ts2 += smem[8 + i]; }
  float mu   = ts / (float)CDIM;
  float var  = ts2 / (float)CDIM - mu * mu;
  float rstd = rsqrtf(var + 1e-5f);
  unsigned short* orow = out + (size_t)row * CDIM;
  for (int c = threadIdx.x; c < CDIM; c += 256)
    orow[c] = f32_to_bf16((xr[c] - mu) * rstd * g[c] + b[c]);
}

// ------------------------------------------- weight convert: f32[K,N]->bf16[N,K]
__global__ void transpose_bf16_kernel(const float* __restrict__ in,
                                      unsigned short* __restrict__ out,
                                      int K, int N) {
  __shared__ float tile[32][33];
  int n0 = blockIdx.x * 32, k0 = blockIdx.y * 32;
  int tx = threadIdx.x & 31, ty = threadIdx.x >> 5;   // ty 0..7
  for (int i = ty; i < 32; i += 8) {
    int k = k0 + i, n = n0 + tx;
    tile[i][tx] = (k < K && n < N) ? in[(size_t)k * N + n] : 0.f;
  }
  __syncthreads();
  for (int i = ty; i < 32; i += 8) {
    int n = n0 + i, k = k0 + tx;
    if (n < N && k < K) out[(size_t)n * K + k] = f32_to_bf16(tile[tx][i]);
  }
}

// ---------------------------------------------------------------- bf16 GEMM
// C[M,N] = A[M,K] * Wt[N,K]^T (+bias)(+residual)(+relu); 128x128x32 tiles,
// double-buffered async LDS staging, 8 waves x (4x2) 16x16 WMMA tiles each.
__global__ __launch_bounds__(256)
void gemm_kernel(const unsigned short* __restrict__ A,
                 const unsigned short* __restrict__ Wt,
                 const float* __restrict__ bias,
                 const float* __restrict__ resid,
                 float* __restrict__ outf,
                 unsigned short* __restrict__ outb,
                 int M, int N, int K, int relu) {
  constexpr int LDA = 40;                       // padded row stride (halfs)
  __shared__ __align__(16) unsigned short As[2][128 * LDA];
  __shared__ __align__(16) unsigned short Bs[2][128 * LDA];
  const int bm = blockIdx.y * 128, bn = blockIdx.x * 128;
  const int tid = threadIdx.x, lane = tid & 31, wave = tid >> 5;
  const int wm = (wave & 1) * 64, wn = (wave >> 1) * 32;
  const int ml = lane & 15, hh = lane >> 4;

  // Per-thread staging slots: thread covers rows (ldr, ldr+64), 8-half chunk ldc.
  const int ldr = tid >> 2, ldc = (tid & 3) * 8;
  const unsigned short* aG0 = A  + (size_t)(bm + ldr) * K + ldc;
  const unsigned short* aG1 = A  + (size_t)(bm + ldr + 64) * K + ldc;
  const unsigned short* bG0 = Wt + (size_t)(bn + ldr) * K + ldc;
  const unsigned short* bG1 = Wt + (size_t)(bn + ldr + 64) * K + ldc;
  const bool fullN = (bn + 128 <= N);
  // OOB B rows are simply not loaded: stale LDS only feeds output columns
  // >= N, which the epilogue discards (D[:,n] depends only on B[:,n]).
  const bool bOk0 = fullN || (bn + ldr) < N;
  const bool bOk1 = fullN || (bn + ldr + 64) < N;

  auto stage = [&](int buf, int k0) {
    cp_async16(aG0 + k0, &As[buf][ldr * LDA + ldc]);
    cp_async16(aG1 + k0, &As[buf][(ldr + 64) * LDA + ldc]);
    if (bOk0) cp_async16(bG0 + k0, &Bs[buf][ldr * LDA + ldc]);
    if (bOk1) cp_async16(bG1 + k0, &Bs[buf][(ldr + 64) * LDA + ldc]);
  };

  v8f acc[4][2];
  for (int mt = 0; mt < 4; mt++)
    for (int nt = 0; nt < 2; nt++)
      for (int r = 0; r < 8; r++) acc[mt][nt][r] = 0.f;

  stage(0, 0);
  const int nk = K / 32;
  for (int kt = 0; kt < nk; kt++) {
    async_wait_all();            // my async fills for buf[kt&1] landed
    __syncthreads();             // everyone's landed; prev buffer free
    if (kt + 1 < nk) stage((kt + 1) & 1, (kt + 1) * 32);

    const unsigned short* Ab = As[kt & 1];
    const unsigned short* Bb = Bs[kt & 1];
    FragBF afr[4], bfr[2];
    for (int mt = 0; mt < 4; mt++) {
      const unsigned short* p = Ab + (wm + mt * 16 + ml) * LDA;
      afr[mt].i.lo = *(const ivec4*)(p + 8 * hh);       // K = 8h..8h+7
      afr[mt].i.hi = *(const ivec4*)(p + 16 + 8 * hh);  // K = 16+8h..
    }
    for (int nt = 0; nt < 2; nt++) {
      const unsigned short* p = Bb + (wn + nt * 16 + ml) * LDA;
      bfr[nt].i.lo = *(const ivec4*)(p + 8 * hh);       // K = 8h..8h+7
      bfr[nt].i.hi = *(const ivec4*)(p + 16 + 8 * hh);  // K = 16+8h..
    }
    for (int mt = 0; mt < 4; mt++)
      for (int nt = 0; nt < 2; nt++)
        acc[mt][nt] = wmma_bf16(afr[mt].v, bfr[nt].v, acc[mt][nt]);
  }

  for (int mt = 0; mt < 4; mt++)
    for (int nt = 0; nt < 2; nt++) {
      int col = bn + wn + nt * 16 + ml;
      if (col >= N) continue;
      float bv = bias ? bias[col] : 0.f;
      for (int r = 0; r < 8; r++) {
        int row = bm + wm + mt * 16 + r + 8 * hh;
        float v = acc[mt][nt][r] + bv;
        size_t idx = (size_t)row * N + col;
        if (resid) v += resid[idx];
        if (relu) v = fmaxf(v, 0.f);
        if (outf) outf[idx] = v;
        if (outb) outb[idx] = f32_to_bf16(v);
      }
    }
}

// ------------------------------------------------------- flash attention
// grid (T/128, H, B); 8 waves, each owns 16 query rows; key tiles of 32.
__global__ __launch_bounds__(256)
void attn_kernel(const unsigned short* __restrict__ q,
                 const unsigned short* __restrict__ k,
                 const unsigned short* __restrict__ v,
                 unsigned short* __restrict__ o) {
  constexpr int LDK = 72, LDV = 40, LDP = 40;
  __shared__ __align__(16) unsigned short Kt[32 * LDK];   // [key][hs]
  __shared__ __align__(16) unsigned short Vt[HSZ * LDV];  // [hs][key]
  __shared__ __align__(16) unsigned short Ps[8][16 * LDP];
  const int qb = blockIdx.x, head = blockIdx.y, batch = blockIdx.z;
  const int tid = threadIdx.x, lane = tid & 31, wave = tid >> 5;
  const int ml = lane & 15, hh = lane >> 4;
  const int q0 = qb * 128 + wave * 16;

  FragBF qf[2];
  {
    size_t base = (size_t)(batch * TT + q0 + ml) * CDIM + head * HSZ;
    for (int c = 0; c < 2; c++) {
      qf[c].i.lo = *(const ivec4*)(q + base + 32 * c + 8 * hh);
      qf[c].i.hi = *(const ivec4*)(q + base + 32 * c + 16 + 8 * hh);
    }
  }

  float mrow[8], lrow[8];
  v8f oacc[4];
  for (int r = 0; r < 8; r++) { mrow[r] = -1e30f; lrow[r] = 0.f; }
  for (int ht = 0; ht < 4; ht++)
    for (int r = 0; r < 8; r++) oacc[ht][r] = 0.f;

  const int nkb = qb * 4 + 4;                 // causal: keys <= qb*128+127
  const int kb_last = (q0 + 15) >> 5;         // last tile visible to this wave
  const int lkey = tid >> 3, lc8 = (tid & 7) * 8;

  for (int kb = 0; kb < nkb; kb++) {
    __syncthreads();                          // prev tile fully consumed
    {
      size_t g = (size_t)(batch * TT + kb * 32 + lkey) * CDIM + head * HSZ + lc8;
      cp_async16(k + g, Kt + lkey * LDK + lc8);      // async K row chunk
      union { ivec4 q4; unsigned short h[8]; } u;    // V: transpose via regs
      u.q4 = *(const ivec4*)(v + g);
      for (int j = 0; j < 8; j++) Vt[(lc8 + j) * LDV + lkey] = u.h[j];
    }
    async_wait_all();
    __syncthreads();
    if (kb > kb_last) continue;               // fully masked for this wave

    v8f s[2];
    for (int t = 0; t < 2; t++) for (int r = 0; r < 8; r++) s[t][r] = 0.f;
    for (int t = 0; t < 2; t++) {
      for (int c = 0; c < 2; c++) {
        FragBF kf;
        const unsigned short* p = Kt + (16 * t + ml) * LDK + 32 * c;
        kf.i.lo = *(const ivec4*)(p + 8 * hh);
        kf.i.hi = *(const ivec4*)(p + 16 + 8 * hh);
        s[t] = wmma_bf16(qf[c].v, kf.v, s[t]);
      }
    }

    const bool boundary = (kb * 32 + 31 > q0);
    for (int t = 0; t < 2; t++)
      for (int r = 0; r < 8; r++) {
        float val = s[t][r] * 0.125f;         // HS^-0.5
        if (boundary) {
          int key = kb * 32 + 16 * t + ml;
          int qi  = q0 + r + 8 * hh;
          if (key > qi) val = -1e30f;
        }
        s[t][r] = val;
      }

    float alpha[8];
    for (int r = 0; r < 8; r++) {
      float mx = fmaxf(s[0][r], s[1][r]);
      for (int d = 8; d >= 1; d >>= 1) mx = fmaxf(mx, __shfl_xor(mx, d, 16));
      float nm = fmaxf(mrow[r], mx);
      float al = __expf(mrow[r] - nm);
      float p0 = __expf(s[0][r] - nm), p1 = __expf(s[1][r] - nm);
      float ps = p0 + p1;
      for (int d = 8; d >= 1; d >>= 1) ps += __shfl_xor(ps, d, 16);
      lrow[r] = lrow[r] * al + ps;
      mrow[r] = nm; alpha[r] = al;
      s[0][r] = p0; s[1][r] = p1;
    }

    // C-layout -> A-layout relayout of P through per-wave LDS.
    for (int t = 0; t < 2; t++)
      for (int r = 0; r < 8; r++)
        Ps[wave][(r + 8 * hh) * LDP + 16 * t + ml] = f32_to_bf16(s[t][r]);
    asm volatile("s_wait_dscnt 0" ::: "memory");
    FragBF pf;
    pf.i.lo = *(const ivec4*)(&Ps[wave][ml * LDP + 8 * hh]);
    pf.i.hi = *(const ivec4*)(&Ps[wave][ml * LDP + 16 + 8 * hh]);

    for (int ht = 0; ht < 4; ht++) {
      for (int r = 0; r < 8; r++) oacc[ht][r] *= alpha[r];
      FragBF vf;
      const unsigned short* p = Vt + (16 * ht + ml) * LDV;
      vf.i.lo = *(const ivec4*)(p + 8 * hh);
      vf.i.hi = *(const ivec4*)(p + 16 + 8 * hh);
      oacc[ht] = wmma_bf16(pf.v, vf.v, oacc[ht]);
    }
  }

  for (int ht = 0; ht < 4; ht++)
    for (int r = 0; r < 8; r++) {
      size_t row = (size_t)(batch * TT + q0 + r + 8 * hh);
      o[row * CDIM + head * HSZ + 16 * ht + ml] =
          f32_to_bf16(oacc[ht][r] / lrow[r]);
    }
}

// ------------------------------------------------------------------- loss
__global__ void loss_kernel(const float* __restrict__ logits,
                            const int* __restrict__ targets,
                            float* __restrict__ loss_out) {
  __shared__ float sm[256], sl[256];
  int row = blockIdx.x;
  const float* lr = logits + (size_t)row * VOCAB;
  float m = -1e30f, l = 0.f;
  for (int c = threadIdx.x; c < VOCAB; c += 256) {
    float x = lr[c];
    if (x > m) { l = l * __expf(m - x) + 1.f; m = x; }
    else       { l += __expf(x - m); }
  }
  sm[threadIdx.x] = m; sl[threadIdx.x] = l;
  __syncthreads();
  for (int s = 128; s >= 1; s >>= 1) {
    if (threadIdx.x < s) {
      float m1 = sm[threadIdx.x], l1 = sl[threadIdx.x];
      float m2 = sm[threadIdx.x + s], l2 = sl[threadIdx.x + s];
      float nm = fmaxf(m1, m2);
      sm[threadIdx.x] = nm;
      sl[threadIdx.x] = l1 * __expf(m1 - nm) + l2 * __expf(m2 - nm);
    }
    __syncthreads();
  }
  if (threadIdx.x == 0) {
    float lse = sm[0] + __logf(sl[0]);
    atomicAdd(loss_out, (lse - lr[targets[row]]) / (float)BT);
  }
}

// ------------------------------------------------------------------- host
extern "C" void kernel_launch(void* const* d_in, const int* in_sizes, int n_in,
                              void* d_out, int out_size, void* d_ws,
                              size_t ws_size, hipStream_t stream) {
  (void)in_sizes; (void)n_in; (void)out_size; (void)ws_size;
  const int*   inputs  = (const int*)d_in[0];
  const int*   targets = (const int*)d_in[1];
  const float* tok_emb = (const float*)d_in[2];
  const float* pos_emb = (const float*)d_in[3];
  const float* ln1_g   = (const float*)d_in[4];
  const float* ln1_b   = (const float*)d_in[5];
  const float* ln2_g   = (const float*)d_in[6];
  const float* ln2_b   = (const float*)d_in[7];
  const float* wq      = (const float*)d_in[8];
  const float* wk      = (const float*)d_in[9];
  const float* wv      = (const float*)d_in[10];
  const float* wo      = (const float*)d_in[11];
  const float* bo      = (const float*)d_in[12];
  const float* w1      = (const float*)d_in[13];
  const float* b1      = (const float*)d_in[14];
  const float* w2      = (const float*)d_in[15];
  const float* b2      = (const float*)d_in[16];
  const float* lnf_g   = (const float*)d_in[17];
  const float* lnf_b   = (const float*)d_in[18];
  const float* w_head  = (const float*)d_in[19];
  const float* b_head  = (const float*)d_in[20];

  float* logits = (float*)d_out;
  float* loss   = logits + (size_t)BT * VOCAB;

  char* ws = (char*)d_ws;
  size_t off = 0;
  auto alloc = [&](size_t bytes) -> void* {
    void* p = ws + off;
    off += (bytes + 255) & ~(size_t)255;
    return p;
  };
  float*          X    = (float*)alloc((size_t)BT * CDIM * 4);
  unsigned short* Abuf = (unsigned short*)alloc((size_t)BT * CDIM * 2);
  unsigned short* Qb   = (unsigned short*)alloc((size_t)BT * CDIM * 2);
  unsigned short* Kb   = (unsigned short*)alloc((size_t)BT * CDIM * 2);
  unsigned short* Vb   = (unsigned short*)alloc((size_t)BT * CDIM * 2);
  unsigned short* Ob   = (unsigned short*)alloc((size_t)BT * CDIM * 2);
  unsigned short* Hid  = (unsigned short*)alloc((size_t)BT * FDIM * 2);
  const size_t CC = (size_t)CDIM * CDIM, CF = (size_t)CDIM * FDIM;
  unsigned short* WT   = (unsigned short*)alloc((4 * CC + 2 * CF) * 2);
  unsigned short* WHT  = (unsigned short*)alloc((size_t)VOCAB * CDIM * 2);
  const size_t WL_Q = 0, WL_K = CC, WL_V = 2 * CC, WL_O = 3 * CC;
  const size_t WL_1 = 4 * CC, WL_2 = 4 * CC + CF;

  const dim3 blk(256);
  const dim3 gCC((CDIM + 31) / 32, (CDIM + 31) / 32);
  const dim3 gCF((FDIM + 31) / 32, (CDIM + 31) / 32);
  const dim3 gFC((CDIM + 31) / 32, (FDIM + 31) / 32);
  auto ggemm = [](int N) { return dim3((N + 127) / 128, BT / 128); };

  // Head weight transposed/converted once (stream-ordered, deterministic).
  transpose_bf16_kernel<<<dim3((VOCAB + 31) / 32, (CDIM + 31) / 32), blk, 0,
                          stream>>>(w_head, WHT, CDIM, VOCAB);

  embed_kernel<<<BT, blk, 0, stream>>>(inputs, tok_emb, pos_emb, X);

  for (int l = 0; l < LNUM; l++) {
    // Convert this layer's weights (transposed, bf16).
    transpose_bf16_kernel<<<gCC, blk, 0, stream>>>(wq + l * CC, WT + WL_Q, CDIM, CDIM);
    transpose_bf16_kernel<<<gCC, blk, 0, stream>>>(wk + l * CC, WT + WL_K, CDIM, CDIM);
    transpose_bf16_kernel<<<gCC, blk, 0, stream>>>(wv + l * CC, WT + WL_V, CDIM, CDIM);
    transpose_bf16_kernel<<<gCC, blk, 0, stream>>>(wo + l * CC, WT + WL_O, CDIM, CDIM);
    transpose_bf16_kernel<<<gCF, blk, 0, stream>>>(w1 + l * CF, WT + WL_1, CDIM, FDIM);
    transpose_bf16_kernel<<<gFC, blk, 0, stream>>>(w2 + l * CF, WT + WL_2, FDIM, CDIM);

    ln_kernel<<<BT, blk, 0, stream>>>(X, ln1_g + (size_t)l * CDIM,
                                      ln1_b + (size_t)l * CDIM, Abuf);
    gemm_kernel<<<ggemm(CDIM), blk, 0, stream>>>(
        Abuf, WT + WL_Q, nullptr, nullptr, nullptr, Qb, BT, CDIM, CDIM, 0);
    gemm_kernel<<<ggemm(CDIM), blk, 0, stream>>>(
        Abuf, WT + WL_K, nullptr, nullptr, nullptr, Kb, BT, CDIM, CDIM, 0);
    gemm_kernel<<<ggemm(CDIM), blk, 0, stream>>>(
        Abuf, WT + WL_V, nullptr, nullptr, nullptr, Vb, BT, CDIM, CDIM, 0);

    attn_kernel<<<dim3(TT / 128, NH, BB), blk, 0, stream>>>(Qb, Kb, Vb, Ob);

    gemm_kernel<<<ggemm(CDIM), blk, 0, stream>>>(
        Ob, WT + WL_O, bo + (size_t)l * CDIM, X, X, nullptr, BT, CDIM, CDIM, 0);

    ln_kernel<<<BT, blk, 0, stream>>>(X, ln2_g + (size_t)l * CDIM,
                                      ln2_b + (size_t)l * CDIM, Abuf);
    gemm_kernel<<<ggemm(FDIM), blk, 0, stream>>>(
        Abuf, WT + WL_1, b1 + (size_t)l * FDIM, nullptr, nullptr, Hid,
        BT, FDIM, CDIM, 1);
    gemm_kernel<<<ggemm(CDIM), blk, 0, stream>>>(
        Hid, WT + WL_2, b2 + (size_t)l * CDIM, X, X, nullptr, BT, CDIM, FDIM, 0);
  }

  ln_kernel<<<BT, blk, 0, stream>>>(X, lnf_g, lnf_b, Abuf);
  gemm_kernel<<<ggemm(VOCAB), blk, 0, stream>>>(
      Abuf, WHT, b_head, nullptr, logits, nullptr, BT, VOCAB, CDIM, 0);

  (void)hipMemsetAsync(loss, 0, sizeof(float), stream);
  loss_kernel<<<BT, blk, 0, stream>>>(logits, targets, loss);
}
